// RNN_decoder_model_32083405701736
// MI455X (gfx1250) — matrix operand order repaired and checked by reading, not compiled, not run on hardware
//
#include <hip/hip_runtime.h>
#include <hip/hip_bf16.h>

// ---------------------------------------------------------------------------
// RNN greedy decoder on gfx1250: bf16 WMMA GEMMs, f32 accumulate.
// Each GEMM wave computes a 16x64 strip (4 independent WMMA accumulators,
// shared A tile) to keep the matrix pipe busy without RAW stalls.
// Stream-ordered kernel chain provides the sequential dependency across the
// 64 decode steps (graph-capture safe: no malloc/sync inside).
// ---------------------------------------------------------------------------

typedef __bf16        v16bf __attribute__((ext_vector_type(16)));
typedef float         v8f   __attribute__((ext_vector_type(8)));
typedef unsigned int  u32x4 __attribute__((ext_vector_type(4)));

static constexpr int V = 32000;   // vocab
static constexpr int E = 128;     // embedding / bottleneck
static constexpr int H = 1024;    // hidden
static constexpr int L = 4;       // layers
static constexpr int B = 32;      // batch
static constexpr int T = 64;      // steps
static constexpr int G = 4 * H;   // 4096 gates

#define DEV static __device__ __forceinline__

DEV unsigned short f2bf(float f) {
  unsigned int u = __float_as_uint(f);
  u += 0x7fffu + ((u >> 16) & 1u);         // round-to-nearest-even
  return (unsigned short)(u >> 16);
}

union Pack32 { u32x4 u[2]; v16bf v; };

// A-matrix tile (16x32 bf16), ISA 7.12.2 layout:
// lanes 0-15: m=lane, k in [0..7] (VGPR0-3) and [16..23] (VGPR4-7)
// lanes 16-31: m=lane-16, k in [8..15] and [24..31]
DEV v16bf loadA(const unsigned short* A, int lda, int m0, int kk) {
  int lane = threadIdx.x & 31;
  int hf   = lane >> 4;
  const unsigned short* row = A + (size_t)(m0 + (lane & 15)) * lda + kk + hf * 8;
  Pack32 r;
  r.u[0] = *(const u32x4*)(row);        // k = hf*8 .. +7
  r.u[1] = *(const u32x4*)(row + 16);   // k = 16 + hf*8 .. +7
  return r.v;
}

// B-matrix tile (32x16 bf16): lanes 0-15 hold column n=lane, k=0..7,16..23;
// lanes 16-31 hold n=lane-16, k=8..15,24..31.  B[k,n] = W[n,k] (row-major W[N,K])
// -> per lane: two 8-bf16 runs (16 apart) of one weight row.
DEV v16bf loadB(const unsigned short* W, int ldw, int n0, int kk) {
  int lane = threadIdx.x & 31;
  int hf   = lane >> 4;
  const unsigned short* row = W + (size_t)(n0 + (lane & 15)) * ldw + kk + hf * 8;
  Pack32 r;
  r.u[0] = *(const u32x4*)(row);
  r.u[1] = *(const u32x4*)(row + 16);
  return r.v;
}

DEV v8f wmma_bf16(v16bf a, v16bf b, v8f c) {
  return __builtin_amdgcn_wmma_f32_16x16x32_bf16(
      /*neg_a=*/false, a, /*neg_b=*/false, b,
      /*c_mod=*/(short)0, c, /*reuse_a=*/false, /*reuse_b=*/false);
}

// Accumulate a 16x64 strip: D[16, n0..n0+63] += A[m0..m0+15, :K] * W^T.
// 4 independent accumulator chains share each A tile.
DEV void gemm_strip(const unsigned short* Amat, int lda, int m0,
                    const unsigned short* Wmat, int ldw, int n0,
                    int K, v8f acc[4]) {
  for (int kk = 0; kk < K; kk += 32) {
    v16bf a = loadA(Amat, lda, m0, kk);
#pragma unroll
    for (int j = 0; j < 4; ++j) {
      v16bf b = loadB(Wmat, ldw, n0 + 16 * j, kk);
      acc[j] = wmma_bf16(a, b, acc[j]);
    }
  }
}

// ---------------------------------------------------------------------------
// One-time per-launch prep
// ---------------------------------------------------------------------------
__global__ void cvt_bf16_kernel(const float* __restrict__ src,
                                unsigned short* __restrict__ dst, int n) {
  for (int i = blockIdx.x * blockDim.x + threadIdx.x; i < n;
       i += gridDim.x * blockDim.x)
    dst[i] = f2bf(src[i]);
}

__global__ void init_state_kernel(const float* __restrict__ hid,
                                  const float* __restrict__ cell,
                                  unsigned short* __restrict__ h,
                                  float* __restrict__ c, int n) {
  for (int i = blockIdx.x * blockDim.x + threadIdx.x; i < n;
       i += gridDim.x * blockDim.x) {
    h[i] = f2bf(hid[i]);
    c[i] = cell[i];
  }
}

__global__ void init_prev_kernel(const int* __restrict__ eos,
                                 int* __restrict__ prev) {
  if (threadIdx.x < B) prev[threadIdx.x] = eos[0];
}

// ---------------------------------------------------------------------------
// Per-step kernels
// ---------------------------------------------------------------------------

// x0[b,e] = emb_W[e, prev[b]] + emb_b[e]   (gather column of [E,V] fp32 -> bf16)
__global__ __launch_bounds__(256)
void embed_kernel(const float* __restrict__ embW, const float* __restrict__ embB,
                  const int* __restrict__ prev, unsigned short* __restrict__ x0) {
  int idx = blockIdx.x * 256 + threadIdx.x;      // B*E = 4096 threads
  int b = idx >> 7, e = idx & 127;
  x0[idx] = f2bf(embW[(size_t)e * V + prev[b]] + embB[e]);
}

// gates[32,4096] = x[32,Kx]@Wih^T + h[32,H]@Whh^T + b_ih + b_hh
// 64 N-strips (of 64) x 2 M-tiles = 128 waves; 8 waves/block -> 16 blocks.
__global__ __launch_bounds__(256)
void lstm_gates_kernel(const unsigned short* __restrict__ x, int Kx,
                       const unsigned short* __restrict__ Wih,
                       const unsigned short* __restrict__ h,
                       const unsigned short* __restrict__ Whh,
                       const float* __restrict__ bih,
                       const float* __restrict__ bhh,
                       float* __restrict__ gates) {
  int tile = blockIdx.x * 8 + (threadIdx.x >> 5);
  int n0 = (tile & 63) * 64;
  int m0 = (tile >> 6) * 16;
  v8f acc[4] = {};
  gemm_strip(x, Kx, m0, Wih, Kx, n0, Kx, acc);
  gemm_strip(h, H, m0, Whh, H, n0, H, acc);
  int lane = threadIdx.x & 31;
  int mb = m0 + ((lane >> 4) << 3);
#pragma unroll
  for (int j = 0; j < 4; ++j) {
    int n = n0 + 16 * j + (lane & 15);
    float bias = bih[n] + bhh[n];
#pragma unroll
    for (int r = 0; r < 8; ++r)
      gates[(size_t)(mb + r) * G + n] = acc[j][r] + bias;
  }
}

// c,h update.  c kept fp32 (in place), h written bf16 (feeds next layer/step).
__global__ __launch_bounds__(256)
void lstm_pointwise_kernel(const float* __restrict__ gates,
                           float* __restrict__ c,
                           unsigned short* __restrict__ h) {
  int idx = blockIdx.x * 256 + threadIdx.x;      // B*H = 32768
  int b = idx >> 10, j = idx & 1023;
  const float* g4 = gates + (size_t)b * G;
  float i_ = g4[j], f_ = g4[H + j], g_ = g4[2 * H + j], o_ = g4[3 * H + j];
  float si = 1.f / (1.f + expf(-i_));
  float sf = 1.f / (1.f + expf(-f_));
  float so = 1.f / (1.f + expf(-o_));
  float cn = sf * c[idx] + si * tanhf(g_);
  float hn = so * tanhf(cn);
  c[idx] = cn;
  h[idx] = f2bf(hn);
}

// bott[32,128] = h3[32,1024]@Wb^T + bb  (bf16 out).
// 2 N-strips x 2 M-tiles = 4 waves -> 1 block of 128 threads.
__global__ __launch_bounds__(128)
void bottleneck_kernel(const unsigned short* __restrict__ h3,
                       const unsigned short* __restrict__ Wb,
                       const float* __restrict__ bb,
                       unsigned short* __restrict__ bott) {
  int tile = threadIdx.x >> 5;                   // 0..3
  int n0 = (tile & 1) * 64;
  int m0 = (tile >> 1) * 16;
  v8f acc[4] = {};
  gemm_strip(h3, H, m0, Wb, H, n0, H, acc);
  int lane = threadIdx.x & 31;
  int mb = m0 + ((lane >> 4) << 3);
#pragma unroll
  for (int j = 0; j < 4; ++j) {
    int n = n0 + 16 * j + (lane & 15);
    float bias = bb[n];
#pragma unroll
    for (int r = 0; r < 8; ++r)
      bott[(mb + r) * E + n] = f2bf(acc[j][r] + bias);
  }
}

// logits[32,32000] = bott[32,128]@Wl^T + bl.
// 500 N-strips x 2 M-tiles = 1000 waves -> 125 blocks.
__global__ __launch_bounds__(256)
void logits_kernel(const unsigned short* __restrict__ bott,
                   const unsigned short* __restrict__ Wl,
                   const float* __restrict__ bl,
                   float* __restrict__ logits) {
  int tile = blockIdx.x * 8 + (threadIdx.x >> 5);
  int n0 = (tile % 500) * 64;
  int m0 = (tile / 500) * 16;
  v8f acc[4] = {};
  gemm_strip(bott, E, m0, Wl, E, n0, E, acc);
  int lane = threadIdx.x & 31;
  int mb = m0 + ((lane >> 4) << 3);
#pragma unroll
  for (int j = 0; j < 4; ++j) {
    int n = n0 + 16 * j + (lane & 15);
    float bias = bl[n];
#pragma unroll
    for (int r = 0; r < 8; ++r)
      logits[(size_t)(mb + r) * V + n] = acc[j][r] + bias;
  }
}

// Per-row softmax + greedy argmax feedback.  One block per batch row.
__global__ __launch_bounds__(256)
void softmax_argmax_kernel(const float* __restrict__ logits,
                           float* __restrict__ probs,
                           int* __restrict__ prev) {
  __shared__ float sred[256];
  __shared__ int   sidx[256];
  int b = blockIdx.x, t = threadIdx.x;
  const float* row = logits + (size_t)b * V;
  float mx = -3.4e38f; int arg = 0;
  for (int i = t; i < V; i += 256) {
    float v = row[i];
    if (v > mx) { mx = v; arg = i; }
  }
  sred[t] = mx; sidx[t] = arg; __syncthreads();
  for (int s = 128; s > 0; s >>= 1) {
    if (t < s) {
      float o = sred[t + s]; int oi = sidx[t + s];
      if (o > sred[t] || (o == sred[t] && oi < sidx[t])) {
        sred[t] = o; sidx[t] = oi;
      }
    }
    __syncthreads();
  }
  float gmax = sred[0]; int garg = sidx[0];
  __syncthreads();
  float sum = 0.f;
  for (int i = t; i < V; i += 256) sum += expf(row[i] - gmax);
  sred[t] = sum; __syncthreads();
  for (int s = 128; s > 0; s >>= 1) {
    if (t < s) sred[t] += sred[t + s];
    __syncthreads();
  }
  float inv = 1.f / sred[0];
  for (int i = t; i < V; i += 256)
    probs[(size_t)b * V + i] = expf(row[i] - gmax) * inv;
  if (t == 0) prev[b] = garg;
}

// ---------------------------------------------------------------------------
// Workspace layout
// ---------------------------------------------------------------------------
constexpr size_t al256(size_t x) { return (x + 255) & ~size_t(255); }

constexpr size_t OFF_WIH0   = 0;
constexpr size_t OFF_WIHR   = al256(OFF_WIH0   + (size_t)G * E * 2);
constexpr size_t OFF_WHH    = al256(OFF_WIHR   + (size_t)(L - 1) * G * H * 2);
constexpr size_t OFF_WB     = al256(OFF_WHH    + (size_t)L * G * H * 2);
constexpr size_t OFF_WL     = al256(OFF_WB     + (size_t)E * H * 2);
constexpr size_t OFF_H      = al256(OFF_WL     + (size_t)V * E * 2);
constexpr size_t OFF_C      = al256(OFF_H      + (size_t)L * B * H * 2);
constexpr size_t OFF_X0     = al256(OFF_C      + (size_t)L * B * H * 4);
constexpr size_t OFF_GATES  = al256(OFF_X0     + (size_t)B * E * 2);
constexpr size_t OFF_BOTT   = al256(OFF_GATES  + (size_t)B * G * 4);
constexpr size_t OFF_LOGITS = al256(OFF_BOTT   + (size_t)B * E * 2);
constexpr size_t OFF_PREV   = al256(OFF_LOGITS + (size_t)B * V * 4);

extern "C" void kernel_launch(void* const* d_in, const int* in_sizes, int n_in,
                              void* d_out, int out_size, void* d_ws, size_t ws_size,
                              hipStream_t stream) {
  (void)in_sizes; (void)n_in; (void)out_size; (void)ws_size;

  const float* hidden  = (const float*)d_in[0];
  const float* cell    = (const float*)d_in[1];
  const float* emb_W   = (const float*)d_in[2];
  const float* emb_b   = (const float*)d_in[3];
  const float* W_ih0   = (const float*)d_in[4];
  const float* W_ihr   = (const float*)d_in[5];
  const float* W_hh    = (const float*)d_in[6];
  const float* b_ih    = (const float*)d_in[7];
  const float* b_hh    = (const float*)d_in[8];
  const float* Wb      = (const float*)d_in[9];
  const float* bb      = (const float*)d_in[10];
  const float* Wl      = (const float*)d_in[11];
  const float* bl      = (const float*)d_in[12];
  const int*   eos     = (const int*)d_in[13];

  float* out = (float*)d_out;
  char*  ws  = (char*)d_ws;

  unsigned short* wih0 = (unsigned short*)(ws + OFF_WIH0);
  unsigned short* wihr = (unsigned short*)(ws + OFF_WIHR);
  unsigned short* whh  = (unsigned short*)(ws + OFF_WHH);
  unsigned short* wb   = (unsigned short*)(ws + OFF_WB);
  unsigned short* wl   = (unsigned short*)(ws + OFF_WL);
  unsigned short* hbf  = (unsigned short*)(ws + OFF_H);
  float*          cbuf = (float*)         (ws + OFF_C);
  unsigned short* x0   = (unsigned short*)(ws + OFF_X0);
  float*          gbuf = (float*)         (ws + OFF_GATES);
  unsigned short* bott = (unsigned short*)(ws + OFF_BOTT);
  float*          lgts = (float*)         (ws + OFF_LOGITS);
  int*            prev = (int*)           (ws + OFF_PREV);

  // One-time: convert weights to bf16; init recurrent state; init feedback.
  cvt_bf16_kernel<<<512,  256, 0, stream>>>(W_ih0, wih0, G * E);
  cvt_bf16_kernel<<<2048, 256, 0, stream>>>(W_ihr, wihr, (L - 1) * G * H);
  cvt_bf16_kernel<<<2048, 256, 0, stream>>>(W_hh,  whh,  L * G * H);
  cvt_bf16_kernel<<<128,  256, 0, stream>>>(Wb,    wb,   E * H);
  cvt_bf16_kernel<<<2048, 256, 0, stream>>>(Wl,    wl,   V * E);
  init_state_kernel<<<512, 256, 0, stream>>>(hidden, cell, hbf, cbuf, L * B * H);
  init_prev_kernel<<<1, 32, 0, stream>>>(eos, prev);

  // 64 sequential decode steps (stream order carries the dependency).
  for (int t = 0; t < T; ++t) {
    embed_kernel<<<16, 256, 0, stream>>>(emb_W, emb_b, prev, x0);
    for (int l = 0; l < L; ++l) {
      const unsigned short* xin = (l == 0) ? x0 : hbf + (size_t)(l - 1) * B * H;
      int Kx = (l == 0) ? E : H;
      const unsigned short* wih =
          (l == 0) ? wih0 : wihr + (size_t)(l - 1) * G * H;
      lstm_gates_kernel<<<16, 256, 0, stream>>>(
          xin, Kx, wih, hbf + (size_t)l * B * H, whh + (size_t)l * G * H,
          b_ih + (size_t)l * G, b_hh + (size_t)l * G, gbuf);
      lstm_pointwise_kernel<<<128, 256, 0, stream>>>(
          gbuf, cbuf + (size_t)l * B * H, hbf + (size_t)l * B * H);
    }
    bottleneck_kernel<<<1, 128, 0, stream>>>(hbf + (size_t)(L - 1) * B * H,
                                             wb, bb, bott);
    logits_kernel<<<125, 256, 0, stream>>>(bott, wl, bl, lgts);
    softmax_argmax_kernel<<<B, 256, 0, stream>>>(
        lgts, out + (size_t)t * B * V, prev);
  }
}
